// InvariantVectorReadout_14070312861733
// MI455X (gfx1250) — matrix operand order repaired and checked by reading, not compiled, not run on hardware
//
#include <hip/hip_runtime.h>
#include <math.h>

// Problem constants (match reference)
#define BN_TOTAL 8192      // B*N = 4*2048
#define KTOK     32
#define DDIM     256
#define TDIM     256
#define NHEAD    4

typedef __attribute__((ext_vector_type(16))) __bf16 v16bf;
typedef __attribute__((ext_vector_type(8)))  float  v8f;

union Frag { v16bf v; uint4 q[2]; };

__device__ __forceinline__ unsigned short f2bf(float f) {
  union { float f; unsigned u; } c; c.f = f;
  unsigned u = c.u;
  u += 0x7FFFu + ((u >> 16) & 1u);   // round-to-nearest-even
  return (unsigned short)(u >> 16);
}

__device__ __forceinline__ float gelu_tanh(float x) {
  float x3 = x * x * x;
  return 0.5f * x * (1.0f + tanhf(0.7978845608028654f * (x + 0.044715f * x3)));
}

// One workgroup-wide matmul: A = 32x256 bf16 in LDS, B = Wt[n][k] bf16 in global
// (column-major-transposed weight so B fragments are contiguous 32B loads).
// 8 waves, wave w owns M-tile (w&1) and N-tiles (w>>1)*4 .. +3.
// Fragment layouts per CDNA5 ISA 7.12.2 (16-bit A 16x32; B 32x16; C/D 16x16 f32).
__device__ __forceinline__ void mm_32x256x256(const unsigned short* sA,
                                              const unsigned short* __restrict__ Wt,
                                              int w, int l, v8f acc[4]) {
  const int mt   = w & 1;
  const int nb   = (w >> 1) * 64;
  const int lm   = l & 15;
  const int aRow = (mt * 16 + lm) * 256;
  const int aOff = (l < 16) ? 0 : 8;    // A: half-wave K offset (K 0-7/16-23 vs 8-15/24-31)
  const int bOff = (l < 16) ? 0 : 8;    // B: half-wave K offset (K 0-7/16-23 vs 8-15/24-31)
#pragma unroll
  for (int kk = 0; kk < 8; ++kk) {
    Frag a;
    const int klo = kk * 32 + aOff;
    a.q[0] = *(const uint4*)(sA + aRow + klo);        // K klo..klo+7   -> VGPR0..3
    a.q[1] = *(const uint4*)(sA + aRow + klo + 16);   // K klo+16..+23  -> VGPR4..7
    const int kb = kk * 32 + bOff;
#pragma unroll
    for (int t = 0; t < 4; ++t) {
      const int n = nb + t * 16 + lm;
      Frag bm;
      const unsigned short* bp = Wt + n * 256 + kb;   // contiguous K in transposed weight
      bm.q[0] = *(const uint4*)(bp);
      bm.q[1] = *(const uint4*)(bp + 16);
      acc[t] = __builtin_amdgcn_wmma_f32_16x16x32_bf16(false, a.v, false, bm.v,
                                                       (short)0, acc[t], false, false);
    }
  }
}

// Kernel 0: transpose + cast weights to bf16 once.
// WtEgo[n*256+k]  = bf16(W_ego[k][n])
// WtHln[n*256+k]  = bf16(W_h[k][n])        (LN half, rows 0..255)
// WtHego[n*256+j] = bf16(W_h[256+j][n])    (ego half, rows 256..511)
__global__ void __launch_bounds__(256) prep_w_kernel(const float* __restrict__ W_ego,
                                                     const float* __restrict__ W_h,
                                                     unsigned short* __restrict__ WtEgo,
                                                     unsigned short* __restrict__ WtHln,
                                                     unsigned short* __restrict__ WtHego) {
  const int i = blockIdx.x * 256 + threadIdx.x;   // 0 .. 3*65536-1
  const int which = i >> 16;
  const int j = i & 65535;
  const int n = j >> 8, k = j & 255;
  if (which == 0)      WtEgo[j]  = f2bf(W_ego[k * 256 + n]);
  else if (which == 1) WtHln[j]  = f2bf(W_h[k * 256 + n]);
  else                 WtHego[j] = f2bf(W_h[(256 + k) * 256 + n]);
}

// Kernel 1: per 32 rows of (B*N):
//   tmp  = ego_ctx @ W_ego + b_ego            (bf16 WMMA, f32 acc)
//   egoC = tmp @ W_h_ego + b_h                (bf16 WMMA, f32 acc) -> fp32 workspace
__global__ void __launch_bounds__(256) ego_kernel(const float* __restrict__ ego_ctx,
                                                  const float* __restrict__ b_ego,
                                                  const float* __restrict__ b_h,
                                                  const unsigned short* __restrict__ WtEgo,
                                                  const unsigned short* __restrict__ WtHego,
                                                  float* __restrict__ egoC) {
  __shared__ __align__(16) unsigned short sX[32 * 256];
  const int tid = threadIdx.x;
  const int w = tid >> 5, l = tid & 31;
  const int row0 = blockIdx.x * 32;

  for (int i = tid; i < 32 * 256; i += 256)
    sX[i] = f2bf(ego_ctx[row0 * 256 + i]);
  __syncthreads();

  v8f acc[4] = {};
  mm_32x256x256(sX, WtEgo, w, l, acc);
  __syncthreads();

  const int mt = w & 1, nb = (w >> 1) * 64, lm = l & 15;
  const int mAdd = (l >= 16) ? 8 : 0;
#pragma unroll
  for (int t = 0; t < 4; ++t) {
    const int n = nb + t * 16 + lm;
    const float be = b_ego[n];
#pragma unroll
    for (int rr = 0; rr < 8; ++rr) {
      const int m = mt * 16 + mAdd + rr;
      sX[m * 256 + n] = f2bf(acc[t][rr] + be);
    }
  }
  __syncthreads();

  v8f acc2[4] = {};
  mm_32x256x256(sX, WtHego, w, l, acc2);
#pragma unroll
  for (int t = 0; t < 4; ++t) {
    const int n = nb + t * 16 + lm;
    const float bh = b_h[n];
#pragma unroll
    for (int rr = 0; rr < 8; ++rr) {
      const int m = mt * 16 + mAdd + rr;
      egoC[(size_t)(row0 + m) * 256 + n] = acc2[t][rr] + bh;
    }
  }
}

// Kernel 2: one workgroup per (b,n). LN -> bf16 LDS -> WMMA GEMM -> GELU ->
// logits -> masked softmax over K -> alpha, v_r, v_u.
__global__ void __launch_bounds__(256) main_kernel(const float* __restrict__ inv_tok,
                                                   const float* __restrict__ r_in,
                                                   const float* __restrict__ u_in,
                                                   const unsigned char* __restrict__ mask,
                                                   const float* __restrict__ ln_scale,
                                                   const float* __restrict__ ln_bias,
                                                   const unsigned short* __restrict__ WtHln,
                                                   const float* __restrict__ egoC,
                                                   const float* __restrict__ W_logit,
                                                   const float* __restrict__ b_logit,
                                                   float* __restrict__ out_vr,
                                                   float* __restrict__ out_vu,
                                                   float* __restrict__ out_alpha) {
  __shared__ __align__(16) unsigned short sA[32 * 256];  // LN output, bf16 (16 KB)
  __shared__ float sH[32 * 256];                         // post-GELU h, f32 (32 KB)
  __shared__ float sLg[32 * 4];
  __shared__ float sAl[32 * 4];

  const int bn  = blockIdx.x;
  const int tid = threadIdx.x;
  const int w = tid >> 5, l = tid & 31;
  const float* inv = inv_tok + (size_t)bn * (KTOK * DDIM);

  // ---- Phase 1: LayerNorm (each wave handles 4 rows of 256) ----
  for (int rr = 0; rr < 4; ++rr) {
    const int row = w * 4 + rr;
    float x[8], s = 0.f, q = 0.f;
#pragma unroll
    for (int j = 0; j < 8; ++j) {
      x[j] = inv[row * 256 + l + 32 * j];
      s += x[j]; q += x[j] * x[j];
    }
#pragma unroll
    for (int off = 16; off > 0; off >>= 1) {
      s += __shfl_xor(s, off, 32);
      q += __shfl_xor(q, off, 32);
    }
    const float mean = s * (1.f / 256.f);
    const float var  = q * (1.f / 256.f) - mean * mean;
    const float rs   = rsqrtf(var + 1e-6f);
#pragma unroll
    for (int j = 0; j < 8; ++j) {
      const int c = l + 32 * j;
      sA[row * 256 + c] = f2bf((x[j] - mean) * rs * ln_scale[c] + ln_bias[c]);
    }
  }
  __syncthreads();

  // ---- Phase 2: h = gelu(ln @ W_h_ln + egoC)  (bf16 WMMA, f32 acc) ----
  v8f acc[4] = {};
  mm_32x256x256(sA, WtHln, w, l, acc);
  const int mt = w & 1, nb = (w >> 1) * 64, lm = l & 15;
  const int mAdd = (l >= 16) ? 8 : 0;
#pragma unroll
  for (int t = 0; t < 4; ++t) {
    const int n = nb + t * 16 + lm;
    const float e = egoC[(size_t)bn * 256 + n];
#pragma unroll
    for (int rr = 0; rr < 8; ++rr) {
      const int m = mt * 16 + mAdd + rr;
      sH[m * 256 + n] = gelu_tanh(acc[t][rr] + e);
    }
  }
  __syncthreads();

  // ---- Phase 3: logits = h @ W_logit + b_logit  (32x4, tiny) ----
  if (tid < 128) {
    const int m = tid >> 2, hh = tid & 3;
    float lg = b_logit[hh];
    for (int c = 0; c < 256; ++c)
      lg += sH[m * 256 + c] * W_logit[c * 4 + hh];
    sLg[m * 4 + hh] = lg;
  }
  __syncthreads();

  // ---- Phase 4: masked softmax over K per head ----
  if (tid < 4) {
    const int hh = tid;
    const unsigned char* mk = mask + (size_t)bn * 32;
    float mx = -3.402823466e38f; int any = 0;
    for (int k = 0; k < 32; ++k)
      if (mk[k]) { any = 1; const float v = sLg[k * 4 + hh]; mx = v > mx ? v : mx; }
    float e[32], sum = 0.f;
    for (int k = 0; k < 32; ++k) {
      const float v = (any && mk[k]) ? __expf(sLg[k * 4 + hh] - mx) : 0.f;
      e[k] = v; sum += v;
    }
    const float inv_s = (sum > 0.f) ? 1.f / sum : 0.f;
    for (int k = 0; k < 32; ++k) {
      const float a = e[k] * inv_s;
      sAl[k * 4 + hh] = a;
      out_alpha[(size_t)bn * 128 + k * 4 + hh] = a;
    }
  }
  __syncthreads();

  // ---- Phase 5: v_r, v_u = alpha^T @ {r,u}  (4x32 * 32x3) ----
  if (tid < 24) {
    const int isU = (tid >= 12);
    const int j = tid - (isU ? 12 : 0);
    const int hh = j / 3, d = j % 3;
    const float* vec = (isU ? u_in : r_in) + (size_t)bn * 96;
    float s = 0.f;
    for (int k = 0; k < 32; ++k)
      s += sAl[k * 4 + hh] * vec[k * 3 + d];
    (isU ? out_vu : out_vr)[(size_t)bn * 12 + hh * 3 + d] = s;
  }
}

extern "C" void kernel_launch(void* const* d_in, const int* in_sizes, int n_in,
                              void* d_out, int out_size, void* d_ws, size_t ws_size,
                              hipStream_t stream) {
  (void)in_sizes; (void)n_in; (void)out_size; (void)ws_size;
  const float*         inv_tok  = (const float*)d_in[0];
  const float*         ego_ctx  = (const float*)d_in[1];
  const float*         r_in     = (const float*)d_in[2];
  const float*         u_in     = (const float*)d_in[3];
  const unsigned char* mask     = (const unsigned char*)d_in[4];
  const float*         W_ego    = (const float*)d_in[5];
  const float*         b_ego    = (const float*)d_in[6];
  const float*         ln_scale = (const float*)d_in[7];
  const float*         ln_bias  = (const float*)d_in[8];
  const float*         W_h      = (const float*)d_in[9];
  const float*         b_h      = (const float*)d_in[10];
  const float*         W_logit  = (const float*)d_in[11];
  const float*         b_logit  = (const float*)d_in[12];

  char* ws = (char*)d_ws;
  unsigned short* WtEgo  = (unsigned short*)(ws);                 // 128 KB
  unsigned short* WtHln  = (unsigned short*)(ws + (128 << 10));   // 128 KB
  unsigned short* WtHego = (unsigned short*)(ws + (256 << 10));   // 128 KB
  float*          egoC   = (float*)(ws + (512 << 10));            // 8 MB

  float* out       = (float*)d_out;
  float* out_vr    = out;                       // (B,N,12)
  float* out_vu    = out + BN_TOTAL * 12;       // (B,N,12)
  float* out_alpha = out + BN_TOTAL * 24;       // (B,N,K,4)

  prep_w_kernel<<<768, 256, 0, stream>>>(W_ego, W_h, WtEgo, WtHln, WtHego);
  ego_kernel<<<BN_TOTAL / 32, 256, 0, stream>>>(ego_ctx, b_ego, b_h, WtEgo, WtHego, egoC);
  main_kernel<<<BN_TOTAL, 256, 0, stream>>>(inv_tok, r_in, u_in, mask, ln_scale, ln_bias,
                                            WtHln, egoC, W_logit, b_logit,
                                            out_vr, out_vu, out_alpha);
}
